// FastWeightGluMLPMultihead_9251359556142
// MI455X (gfx1250) — matrix operand order repaired and checked by reading, not compiled, not run on hardware
//
#include <hip/hip_runtime.h>
#include <math.h>

typedef __bf16 bf16;
typedef __attribute__((ext_vector_type(16))) __bf16 v16bf;
typedef __attribute__((ext_vector_type(8)))  __bf16 v8bf;
typedef __attribute__((ext_vector_type(4)))  __bf16 v4bf;
typedef __attribute__((ext_vector_type(8)))  float  v8f;
typedef __attribute__((ext_vector_type(4)))  unsigned int u32x4;
typedef __attribute__((ext_vector_type(8)))  int i32x8;
typedef __attribute__((ext_vector_type(4)))  int i32x4;
typedef unsigned short u16;

#define WMMA_BF16(fa, fb, acc) \
  __builtin_amdgcn_wmma_f32_16x16x32_bf16(false, (fa), false, (fb), (short)0, (acc), false, false)

// Problem constants
static constexpr int   kB = 4, kL = 4096, kDIM = 1024, kHD = 64, kH = 16, kCH = 64, kNC = 64;
static constexpr int   kBH = kB * kH;               // 64 sequences
static constexpr float NS_A = 3.4445f, NS_B = -4.775f, NS_C = 2.0315f;
static constexpr float BASE_LR_INV = -4.6001660f;   // 0.01 + log(-expm1(-0.01))

// Fast reciprocal (v_rcp_f32) — all consumers are bf16, so 1-ulp-ish rcp is fine
// and avoids the v_div_scale/v_fma IEEE division expansion.
__device__ __forceinline__ float fast_rcp(float x) { return __builtin_amdgcn_rcpf(x); }
__device__ __forceinline__ float sigmoidf_(float x) { return fast_rcp(1.0f + __expf(-x)); }

// ---------------------------------------------------------------------------
// Tensor Data Mover: 8KB (4096 x bf16) contiguous global -> LDS copy.
// D# built per CDNA5 ISA (count=1, data_size=2B,
// 1-D tile: tile_dim0 = 4096, tensor_dim0 = 4096, type=2 "image").
#if __has_builtin(__builtin_amdgcn_tensor_load_to_lds) && \
    __has_builtin(__builtin_amdgcn_s_wait_tensorcnt)
#define HAVE_TDM 1
#endif

typedef __attribute__((address_space(3))) char lds_char;
__device__ __forceinline__ unsigned lds_offset_of(const void* p) {
  return (unsigned)(unsigned long long)(lds_char*)p;
}

__device__ __forceinline__ void tdm_load_8kb(void* lds, const void* g) {
#ifdef HAVE_TDM
  unsigned lds_off = lds_offset_of(lds);
  unsigned long long ga = (unsigned long long)g;
  u32x4 g0 = {1u,                                   // count=1, no gather, user D#
              lds_off,                              // lds_addr (bytes)
              (unsigned)ga,                         // global_addr[31:0]
              (unsigned)((ga >> 32) & 0x01FFFFFFu) | (2u << 30)};  // ga[56:32] | type=2
  i32x8 g1 = {(int)0x00010000,                      // wg_mask=0, data_size=1 (2 bytes)
              (int)(4096u << 16),                   // tensor_dim0[15:0] << 16 (=4096)
              (int)0x00010000,                      // tensor_dim0[31:16]=0 | tensor_dim1=1
              (int)(4096u << 16),                   // tensor_dim1 hi=0 | tile_dim0=4096
              1,                                    // tile_dim1=1, tile_dim2=0
              4096,                                 // tensor_dim0_stride[31:0]
              0, 0};
  i32x4 z4 = {0, 0, 0, 0};
#if defined(__clang_major__) && __clang_major__ >= 23
  i32x8 z8 = {0, 0, 0, 0, 0, 0, 0, 0};
  __builtin_amdgcn_tensor_load_to_lds(g0, g1, z4, z4, z8, 0);
#else
  __builtin_amdgcn_tensor_load_to_lds(g0, g1, z4, z4, 0);
#endif
#else
  (void)lds; (void)g;
#endif
}

// ---------------------------------------------------------------------------
// WMMA fragment loaders (gfx1250 wave32 layouts, ISA 7.12.2).
// A fragment: element (m,k) at p[m*RS + k*CS].  Contiguous case (CS==1) is
// two ds_load_b128 per lane (K = kb..kb+7 and kb+16..kb+23 are contiguous).
template <int RS, int CS>
__device__ __forceinline__ v16bf frag_a(const bf16* p, int lane) {
  int m  = lane & 15;
  int kb = (lane >> 4) ? 8 : 0;
  if constexpr (CS == 1) {
    const bf16* q = p + m * RS + kb;
    v8bf lo = *(const v8bf*)q;
    v8bf hi = *(const v8bf*)(q + 16);
    return __builtin_shufflevector(lo, hi, 0, 1, 2, 3, 4, 5, 6, 7,
                                   8, 9, 10, 11, 12, 13, 14, 15);
  } else {
    v16bf r;
#pragma unroll
    for (int i = 0; i < 8; ++i) {
      int k = ((i < 4) ? (2 * i) : (8 + 2 * i)) + kb;
      r[2 * i]     = p[m * RS + k * CS];
      r[2 * i + 1] = p[m * RS + (k + 1) * CS];
    }
    return r;
  }
}
// B fragment: element (k,n) at p[k*RS + n*CS].  When RS==1 (B stored
// transposed / n-major) the 16 K values per lane are two contiguous 16B runs (kb..kb+7, kb+16..kb+23).
template <int RS, int CS>
__device__ __forceinline__ v16bf frag_b(const bf16* p, int lane) {
  int n  = lane & 15;
  int kb = (lane >> 4) ? 8 : 0;
  if constexpr (RS == 1) {
    const bf16* q = p + n * CS + kb;
    v8bf lo = *(const v8bf*)q;
    v8bf hi = *(const v8bf*)(q + 16);
    return __builtin_shufflevector(lo, hi, 0, 1, 2, 3, 4, 5, 6, 7,
                                   8, 9, 10, 11, 12, 13, 14, 15);
  } else {
    v16bf r;
#pragma unroll
    for (int i = 0; i < 8; ++i) {
      int k = ((i < 4) ? (2 * i) : (8 + 2 * i)) + kb;
      r[2 * i]     = p[k * RS + n * CS];
      r[2 * i + 1] = p[(k + 1) * RS + n * CS];
    }
    return r;
  }
}

// ---------------------------------------------------------------------------
// Generic 64x64x64 bf16 matmul; strides are compile-time so the fragment
// loaders specialize.  256 threads / 8 waves, 2 C-tiles per wave.
template <int ARS, int ACS, int BRS, int BCS>
__device__ void mm64(bf16* __restrict__ D, int ldd,
                     const bf16* __restrict__ A, const bf16* __restrict__ B, int tid) {
  __syncthreads();
  int wave = tid >> 5, lane = tid & 31;
#pragma unroll
  for (int t = 0; t < 2; ++t) {
    int tile = wave * 2 + t;
    int tr = (tile >> 2) * 16, tc = (tile & 3) * 16;
    v8f acc = {};
#pragma unroll
    for (int kk = 0; kk < 64; kk += 32) {
      v16bf fa = frag_a<ARS, ACS>(A + tr * ARS + kk * ACS, lane);
      v16bf fb = frag_b<BRS, BCS>(B + kk * BRS + tc * BCS, lane);
      acc = WMMA_BF16(fa, fb, acc);
    }
    int n = lane & 15, mo = (lane >> 4) * 8;
#pragma unroll
    for (int i = 0; i < 8; ++i)
      D[(tr + mo + i) * ldd + tc + n] = (bf16)acc[i];
  }
  __syncthreads();
}

// ---------------------------------------------------------------------------
// Newton-Schulz quintic, 5 steps, in-place on X (64x64 bf16 in LDS).
__device__ void ns5(bf16* X, bf16* Amat, bf16* Tmat, float* redbuf, float* fro, int tid) {
  float p = 0.f;
#pragma unroll
  for (int i = 0; i < 2; ++i) {
    v8bf xv = *(const v8bf*)(X + (i * 256 + tid) * 8);
#pragma unroll
    for (int j = 0; j < 8; ++j) { float v = (float)xv[j]; p += v * v; }
  }
  redbuf[tid] = p;
  __syncthreads();
  if (tid == 0) {
    float s = 0.f;
    for (int i = 0; i < 256; ++i) s += redbuf[i];
    *fro = fast_rcp(sqrtf(s) + 1e-7f);
  }
  __syncthreads();
  float inv = *fro;
#pragma unroll
  for (int i = 0; i < 2; ++i) {
    int e = (i * 256 + tid) * 8;
    v8bf xv = *(const v8bf*)(X + e);
#pragma unroll
    for (int j = 0; j < 8; ++j) xv[j] = (bf16)((float)xv[j] * inv);
    *(v8bf*)(X + e) = xv;
  }
  __syncthreads();
  for (int it = 0; it < 5; ++it) {
    mm64<64, 1, 1, 64>(Amat, 64, X, X, tid);        // A = X @ X^T   (both vector paths)
    mm64<64, 1, 64, 1>(Tmat, 64, Amat, Amat, tid);  // T = A @ A
#pragma unroll
    for (int i = 0; i < 2; ++i) {
      int e = (i * 256 + tid) * 8;
      v8bf av = *(const v8bf*)(Amat + e);
      v8bf tv = *(const v8bf*)(Tmat + e);
#pragma unroll
      for (int j = 0; j < 8; ++j) av[j] = (bf16)(NS_B * (float)av[j] + NS_C * (float)tv[j]);
      *(v8bf*)(Amat + e) = av;                      // Bm
    }
    __syncthreads();
    mm64<64, 1, 64, 1>(Tmat, 64, Amat, X, tid);     // T = Bm @ X
#pragma unroll
    for (int i = 0; i < 2; ++i) {
      int e = (i * 256 + tid) * 8;
      v8bf xv = *(const v8bf*)(X + e);
      v8bf tv = *(const v8bf*)(Tmat + e);
#pragma unroll
      for (int j = 0; j < 8; ++j) xv[j] = (bf16)(NS_A * (float)xv[j] + (float)tv[j]);
      *(v8bf*)(X + e) = xv;
    }
    __syncthreads();
  }
}

// W += G; renormalize columns to initial column norms wn[].
__device__ void update_w(bf16* W, const bf16* G, const float* wn, float* colred, int tid) {
#pragma unroll
  for (int i = 0; i < 2; ++i) {
    int e = (i * 256 + tid) * 8;
    v8bf wv = *(const v8bf*)(W + e);
    v8bf gv = *(const v8bf*)(G + e);
#pragma unroll
    for (int j = 0; j < 8; ++j) wv[j] = (bf16)((float)wv[j] + (float)gv[j]);
    *(v8bf*)(W + e) = wv;
  }
  __syncthreads();
  if (tid < 64) {
    float s = 0.f;
    for (int r = 0; r < 64; ++r) { float v = (float)W[r * 64 + tid]; s += v * v; }
    colred[tid] = wn[tid] * fast_rcp(sqrtf(s) + 1e-5f);
  }
  __syncthreads();
#pragma unroll
  for (int i = 0; i < 2; ++i) {
    int e = (i * 256 + tid) * 8;
    int cb = e & 63;
    v8bf wv = *(const v8bf*)(W + e);
#pragma unroll
    for (int j = 0; j < 8; ++j) wv[j] = (bf16)((float)wv[j] * colred[cb + j]);
    *(v8bf*)(W + e) = wv;
  }
  __syncthreads();
}

// ---------------------------------------------------------------------------
// K0: f32 -> bf16 conversion (8 elements / thread / iteration)
__global__ void cvt_bf16_kernel(const float* __restrict__ src, u16* __restrict__ dst, long n8) {
  long i = (long)blockIdx.x * 256 + threadIdx.x;
  long stride = (long)gridDim.x * 256;
  v8bf* d = (v8bf*)dst;
  for (; i < n8; i += stride) {
    const float* s = src + i * 8;
    v8bf o;
#pragma unroll
    for (int j = 0; j < 8; ++j) o[j] = (bf16)s[j];
    d[i] = o;
  }
}

// K1: qkv = silu(x @ Wqkv), scattered to q/k/v in [b*h][l][64] bf16 layout.
__global__ __launch_bounds__(256) void gemm_qkv_kernel(const u16* __restrict__ xb_,
                                                       const u16* __restrict__ wb_,
                                                       u16* __restrict__ q_, u16* __restrict__ k_,
                                                       u16* __restrict__ v_) {
  const bf16* xb = (const bf16*)xb_;
  const bf16* wb = (const bf16*)wb_;
  bf16 *qd = (bf16*)q_, *kd = (bf16*)k_, *vd = (bf16*)v_;
  __shared__ __align__(64) bf16 As[128 * 32];     // row-major [m][k]
  __shared__ __align__(64) bf16 BsT[128 * 32];    // transposed  [n][k]
  int tid = threadIdx.x, wave = tid >> 5, lane = tid & 31;
  int m0 = blockIdx.y * 128, n0 = blockIdx.x * 128;
  v8f acc[8] = {};
  for (int k0 = 0; k0 < kDIM; k0 += 32) {
    __syncthreads();
#pragma unroll
    for (int i = 0; i < 2; ++i) {               // A: vector copy (b128)
      int vec = i * 256 + tid;                  // 512 x v8bf
      int row = vec >> 2, col = (vec & 3) * 8;
      *(v8bf*)(As + vec * 8) = *(const v8bf*)(xb + (long)(m0 + row) * kDIM + k0 + col);
    }
#pragma unroll
    for (int i = 0; i < 2; ++i) {               // B: coalesced read, transposed LDS write
      int vec = i * 256 + tid;
      int kk = vec >> 4, nn = (vec & 15) * 8;
      v8bf w = *(const v8bf*)(wb + (long)(k0 + kk) * (3 * kDIM) + n0 + nn);
#pragma unroll
      for (int j = 0; j < 8; ++j) BsT[(nn + j) * 32 + kk] = w[j];
    }
    __syncthreads();
    v16bf fa = frag_a<32, 1>(As + wave * 16 * 32, lane);
#pragma unroll
    for (int t = 0; t < 8; ++t) {
      v16bf fb = frag_b<1, 32>(BsT + t * 16 * 32, lane);
      acc[t] = WMMA_BF16(fa, fb, acc[t]);
    }
  }
  int nl = lane & 15, mo = (lane >> 4) * 8;
#pragma unroll
  for (int t = 0; t < 8; ++t) {
#pragma unroll
    for (int i = 0; i < 8; ++i) {
      int m = m0 + wave * 16 + mo + i;
      int j = n0 + t * 16 + nl;
      float val = acc[t][i];
      val = val * sigmoidf_(val);               // silu (exp + v_rcp, no div chain)
      int comp = j >> 10, r = j & 1023, hh = r >> 6, d = r & 63;
      int b = m >> 12, l = m & 4095;
      long idx = (((long)(b * kH + hh)) * kL + l) * kHD + d;
      bf16* dst = (comp == 0) ? qd : (comp == 1) ? kd : vd;
      dst[idx] = (bf16)val;
    }
  }
}

// K2: row-normalize q and k (rows of length 64)
__global__ void qknorm_kernel(u16* __restrict__ q_, u16* __restrict__ k_) {
  long row = (long)blockIdx.x * blockDim.x + threadIdx.x;
  if (row >= (long)kBH * kL) return;
  bf16* bufs[2] = {(bf16*)q_ + row * kHD, (bf16*)k_ + row * kHD};
#pragma unroll
  for (int w = 0; w < 2; ++w) {
    bf16* p = bufs[w];
    float s = 0.f;
#pragma unroll
    for (int c = 0; c < 8; ++c) {
      v8bf x = *(const v8bf*)(p + c * 8);
#pragma unroll
      for (int j = 0; j < 8; ++j) { float v = (float)x[j]; s += v * v; }
    }
    float inv = fast_rcp(sqrtf(s) + 1e-5f);
#pragma unroll
    for (int c = 0; c < 8; ++c) {
      v8bf x = *(const v8bf*)(p + c * 8);
#pragma unroll
      for (int j = 0; j < 8; ++j) x[j] = (bf16)((float)x[j] * inv);
      *(v8bf*)(p + c * 8) = x;
    }
  }
}

// K3: lr = softplus(x @ Wlr + blr + BASE_LR_INV) -> lrw[comp][bh][l] (f32)
__global__ void lr_kernel(const float* __restrict__ x, const float* __restrict__ Wlr,
                          const float* __restrict__ blr, float* __restrict__ lrw) {
  long gid = (long)blockIdx.x * blockDim.x + threadIdx.x;
  if (gid >= (long)kB * kL * 48) return;
  int j = (int)(gid % 48);
  long t = gid / 48;
  float acc = blr[j] + BASE_LR_INV;
  const float* xr = x + t * kDIM;
  for (int kk = 0; kk < kDIM; ++kk) acc += xr[kk] * Wlr[(long)kk * 48 + j];
  float sp = (acc > 20.f) ? acc : log1pf(__expf(acc));   // softplus
  int comp = j % 3, hh = j / 3;
  int b = (int)(t >> 12), l = (int)(t & 4095);
  lrw[(long)comp * kBH * kL + (long)(b * kH + hh) * kL + l] = sp;
}

// K4: sequential TTT scan, one workgroup per (b,h).
__global__ __launch_bounds__(256) void ttt_scan_kernel(
    const float* __restrict__ w0g, const float* __restrict__ w1g, const float* __restrict__ w2g,
    const u16* __restrict__ q_, const u16* __restrict__ k_, const u16* __restrict__ v_,
    const float* __restrict__ lrw, u16* __restrict__ o_) {
  const bf16* qg = (const bf16*)q_;
  const bf16* kg = (const bf16*)k_;
  const bf16* vg = (const bf16*)v_;
  bf16* og = (bf16*)o_;

  int bh = blockIdx.x;              // 0..63
  int h  = bh & (kH - 1);
  int tid = threadIdx.x;

  __shared__ __align__(64) bf16 W0s[64 * 64], W1s[64 * 64], W2s[64 * 64];
  __shared__ __align__(64) bf16 Ks[64 * 64], Vs[64 * 64], Qs[64 * 64];
  __shared__ __align__(64) bf16 T1[64 * 64], T2[64 * 64], T3[64 * 64], T4[64 * 64];
  __shared__ __align__(64) bf16 N1[64 * 64], N2[64 * 64];
  __shared__ float l0s[64], l1s[64], l2s[64];
  __shared__ float nrm0[64], nrm1[64], nrm2[64], colred[64];
  __shared__ float redbuf[256];
  __shared__ float fro;

  // Load initial fast weights (batch-tiled => weight index is h only).
#pragma unroll
  for (int i = 0; i < 16; ++i) {
    int e = i * 256 + tid;
    W0s[e] = (bf16)w0g[(long)h * 4096 + e];
    W1s[e] = (bf16)w1g[(long)h * 4096 + e];
    W2s[e] = (bf16)w2g[(long)h * 4096 + e];
  }
  __syncthreads();
  // Initial column norms (axis=1 of [bh,d,dh] == rows of row-major [64][64]).
  if (tid < 64) {
    float s0 = 0.f, s1 = 0.f, s2 = 0.f;
    for (int r = 0; r < 64; ++r) {
      float a = (float)W0s[r * 64 + tid]; s0 += a * a;
      float b = (float)W1s[r * 64 + tid]; s1 += b * b;
      float c = (float)W2s[r * 64 + tid]; s2 += c * c;
    }
    nrm0[tid] = sqrtf(s0); nrm1[tid] = sqrtf(s1); nrm2[tid] = sqrtf(s2);
  }
  __syncthreads();

  for (int c = 0; c < kNC; ++c) {
    long base = ((long)bh * kL + (long)c * kCH) * kHD;   // chunk tile base
#ifdef HAVE_TDM
    if (tid < 32) {                       // one wave issues the three TDM copies
      tdm_load_8kb(Ks, kg + base);
      tdm_load_8kb(Vs, vg + base);
      tdm_load_8kb(Qs, qg + base);
      __builtin_amdgcn_s_wait_tensorcnt(0);
    }
#else
#pragma unroll
    for (int i = 0; i < 2; ++i) {
      int vec = i * 256 + tid;
      *(v8bf*)(Ks + vec * 8) = *(const v8bf*)(kg + base + vec * 8);
      *(v8bf*)(Vs + vec * 8) = *(const v8bf*)(vg + base + vec * 8);
      *(v8bf*)(Qs + vec * 8) = *(const v8bf*)(qg + base + vec * 8);
    }
#endif
    if (tid >= 64 && tid < 128) {
      int t64 = tid - 64;
      long lbase = (long)bh * kL + (long)c * kCH + t64;
      l0s[t64] = lrw[lbase];
      l1s[t64] = lrw[(long)kBH * kL + lbase];
      l2s[t64] = lrw[2L * kBH * kL + lbase];
    }
    if (c + 1 < kNC) {  // prefetch next chunk into cache (global_prefetch_b8)
      long nb = base + (long)kCH * kHD;
      __builtin_prefetch(kg + nb + (long)tid * 16, 0, 1);
      __builtin_prefetch(vg + nb + (long)tid * 16, 0, 1);
      __builtin_prefetch(qg + nb + (long)tid * 16, 0, 1);
    }

    // Forward through current fast weights.
    mm64<64, 1, 64, 1>(T1, 64, Ks, W0s, tid);       // gate  = K @ W0
    mm64<64, 1, 64, 1>(T2, 64, Ks, W2s, tid);       // hbm   = K @ W2
    mm64<64, 1, 1, 64>(T4, 64, Vs, W1s, tid);       // dh    = V @ W1^T

    // Elementwise: hidden*l1 -> T3 ; dhbm*l2 -> T2 ; dgba*l0 -> T1
#pragma unroll
    for (int i = 0; i < 2; ++i) {
      int e = (i * 256 + tid) * 8;
      int row = e >> 6;
      float l0v = l0s[row], l1v = l1s[row], l2v = l2s[row];
      v8bf gv = *(const v8bf*)(T1 + e);
      v8bf hbv = *(const v8bf*)(T2 + e);
      v8bf dv = *(const v8bf*)(T4 + e);
      v8bf t1o, t2o, t3o;
#pragma unroll
      for (int j = 0; j < 8; ++j) {
        float g = (float)gv[j], hb = (float)hbv[j], d = (float)dv[j];
        float s = sigmoidf_(g), sg = g * s;
        t3o[j] = (bf16)(sg * hb * l1v);                              // hidden*l1
        t2o[j] = (bf16)(d * sg * l2v);                               // dhbm*l2
        t1o[j] = (bf16)((d * hb) * s * (1.f + g * (1.f - s)) * l0v); // dgba*l0
      }
      *(v8bf*)(T3 + e) = t3o;
      *(v8bf*)(T2 + e) = t2o;
      *(v8bf*)(T1 + e) = t1o;
    }
    __syncthreads();

    // g1 = (hidden*l1)^T @ V  -> NS -> W1 update
    mm64<1, 64, 64, 1>(T4, 64, T3, Vs, tid);
    ns5(T4, N1, N2, redbuf, &fro, tid);
    update_w(W1s, T4, nrm1, colred, tid);

    // g0 = K^T @ (dgba*l0) -> NS -> W0 update
    mm64<1, 64, 64, 1>(T3, 64, Ks, T1, tid);
    ns5(T3, N1, N2, redbuf, &fro, tid);
    update_w(W0s, T3, nrm0, colred, tid);

    // g2 = K^T @ (dhbm*l2) -> NS -> W2 update
    mm64<1, 64, 64, 1>(T3, 64, Ks, T2, tid);
    ns5(T3, N1, N2, redbuf, &fro, tid);
    update_w(W2s, T3, nrm2, colred, tid);

    // Output: o = (silu(Q@W0n) * (Q@W2n)) @ W1n
    mm64<64, 1, 64, 1>(T1, 64, Qs, W0s, tid);
    mm64<64, 1, 64, 1>(T2, 64, Qs, W2s, tid);
#pragma unroll
    for (int i = 0; i < 2; ++i) {
      int e = (i * 256 + tid) * 8;
      v8bf a = *(const v8bf*)(T1 + e);
      v8bf b = *(const v8bf*)(T2 + e);
#pragma unroll
      for (int j = 0; j < 8; ++j) {
        float av = (float)a[j];
        a[j] = (bf16)(av * sigmoidf_(av) * (float)b[j]);
      }
      *(v8bf*)(T1 + e) = a;
    }
    __syncthreads();
    mm64<64, 1, 64, 1>(og + base, 64, T1, W1s, tid);
  }
}

// K5: RMSNorm over dim=1024 (gather heads), apply affine scale, bf16 out.
__global__ __launch_bounds__(256) void rms_kernel(const u16* __restrict__ o_,
                                                  const float* __restrict__ scale,
                                                  u16* __restrict__ on_) {
  const bf16* ob = (const bf16*)o_;
  bf16* on = (bf16*)on_;
  int t = blockIdx.x;                 // token 0..16383
  int b = t >> 12, l = t & 4095;
  __shared__ float red[256];
  int d0 = threadIdx.x * 4;           // 4 contiguous dims per thread (same head)
  int hh = d0 >> 6;
  const bf16* src = ob + (((long)(b * kH + hh)) * kL + l) * kHD + (d0 & 63);
  v4bf vv = *(const v4bf*)src;
  float s = 0.f;
#pragma unroll
  for (int j = 0; j < 4; ++j) { float v = (float)vv[j]; s += v * v; }
  red[threadIdx.x] = s;
  __syncthreads();
  for (int off = 128; off > 0; off >>= 1) {
    if (threadIdx.x < off) red[threadIdx.x] += red[threadIdx.x + off];
    __syncthreads();
  }
  float inv = rsqrtf(red[0] / (float)kDIM + 1e-5f);
  v4bf o4;
#pragma unroll
  for (int j = 0; j < 4; ++j) o4[j] = (bf16)((float)vv[j] * inv * scale[d0 + j]);
  *(v4bf*)(on + (long)t * kDIM + d0) = o4;
}

// K6: out = obn @ Wproj, f32 output.
__global__ __launch_bounds__(256) void gemm_proj_kernel(const u16* __restrict__ a_,
                                                        const u16* __restrict__ wb_,
                                                        float* __restrict__ out) {
  const bf16* ab = (const bf16*)a_;
  const bf16* wb = (const bf16*)wb_;
  __shared__ __align__(64) bf16 As[128 * 32];
  __shared__ __align__(64) bf16 BsT[128 * 32];
  int tid = threadIdx.x, wave = tid >> 5, lane = tid & 31;
  int m0 = blockIdx.y * 128, n0 = blockIdx.x * 128;
  v8f acc[8] = {};
  for (int k0 = 0; k0 < kDIM; k0 += 32) {
    __syncthreads();
#pragma unroll
    for (int i = 0; i < 2; ++i) {
      int vec = i * 256 + tid;
      int row = vec >> 2, col = (vec & 3) * 8;
      *(v8bf*)(As + vec * 8) = *(const v8bf*)(ab + (long)(m0 + row) * kDIM + k0 + col);
    }
#pragma unroll
    for (int i = 0; i < 2; ++i) {
      int vec = i * 256 + tid;
      int kk = vec >> 4, nn = (vec & 15) * 8;
      v8bf w = *(const v8bf*)(wb + (long)(k0 + kk) * kDIM + n0 + nn);
#pragma unroll
      for (int j = 0; j < 8; ++j) BsT[(nn + j) * 32 + kk] = w[j];
    }
    __syncthreads();
    v16bf fa = frag_a<32, 1>(As + wave * 16 * 32, lane);
#pragma unroll
    for (int t = 0; t < 8; ++t) {
      v16bf fb = frag_b<1, 32>(BsT + t * 16 * 32, lane);
      acc[t] = WMMA_BF16(fa, fb, acc[t]);
    }
  }
  int nl = lane & 15, mo = (lane >> 4) * 8;
#pragma unroll
  for (int t = 0; t < 8; ++t)
#pragma unroll
    for (int i = 0; i < 8; ++i) {
      long m = m0 + wave * 16 + mo + i;
      long j = n0 + t * 16 + nl;
      out[m * kDIM + j] = acc[t][i];
    }
}

// ---------------------------------------------------------------------------
extern "C" void kernel_launch(void* const* d_in, const int* in_sizes, int n_in,
                              void* d_out, int out_size, void* d_ws, size_t ws_size,
                              hipStream_t stream) {
  (void)in_sizes; (void)n_in; (void)out_size; (void)ws_size;
  const float* x     = (const float*)d_in[0];
  const float* w0    = (const float*)d_in[1];
  const float* w1    = (const float*)d_in[2];
  const float* w2    = (const float*)d_in[3];
  const float* Wqkv  = (const float*)d_in[4];
  const float* Wlr   = (const float*)d_in[5];
  const float* blr   = (const float*)d_in[6];
  const float* Wproj = (const float*)d_in[7];
  const float* oscal = (const float*)d_in[8];
  float* out = (float*)d_out;

  // Workspace carve-out (256B aligned slices)
  size_t off = 0;
  auto carve = [&](size_t bytes) -> void* {
    void* p = (char*)d_ws + off;
    off += (bytes + 255) & ~(size_t)255;
    return p;
  };
  const long NTOK = (long)kB * kL;                    // 16384
  u16*  xb    = (u16*)carve((size_t)NTOK * kDIM * 2);         // x bf16
  u16*  wqkvb = (u16*)carve((size_t)kDIM * 3 * kDIM * 2);     // Wqkv bf16
  u16*  wprjb = (u16*)carve((size_t)kDIM * kDIM * 2);         // Wproj bf16
  u16*  qb    = (u16*)carve((size_t)kBH * kL * kHD * 2);
  u16*  kb    = (u16*)carve((size_t)kBH * kL * kHD * 2);
  u16*  vb    = (u16*)carve((size_t)kBH * kL * kHD * 2);
  float* lrw  = (float*)carve((size_t)3 * kBH * kL * 4);
  u16*  ob    = (u16*)carve((size_t)kBH * kL * kHD * 2);
  u16*  obn   = (u16*)carve((size_t)NTOK * kDIM * 2);

  // K0: conversions
  cvt_bf16_kernel<<<2048, 256, 0, stream>>>(x, xb, NTOK * kDIM / 8);
  cvt_bf16_kernel<<<1024, 256, 0, stream>>>(Wqkv, wqkvb, (long)kDIM * 3 * kDIM / 8);
  cvt_bf16_kernel<<<512, 256, 0, stream>>>(Wproj, wprjb, (long)kDIM * kDIM / 8);

  // K1: qkv GEMM + silu + head scatter
  gemm_qkv_kernel<<<dim3(3 * kDIM / 128, NTOK / 128), 256, 0, stream>>>(xb, wqkvb, qb, kb, vb);

  // K2: q/k row normalization
  qknorm_kernel<<<(int)((kBH * (long)kL + 255) / 256), 256, 0, stream>>>(qb, kb);

  // K3: learning rates
  lr_kernel<<<(int)((NTOK * 48 + 255) / 256), 256, 0, stream>>>(x, Wlr, blr, lrw);

  // K4: sequential TTT scan (one WG per (b,h))
  ttt_scan_kernel<<<kBH, 256, 0, stream>>>(w0, w1, w2, qb, kb, vb, lrw, ob);

  // K5: RMSNorm + scale
  rms_kernel<<<(int)NTOK, 256, 0, stream>>>(ob, oscal, obn);

  // K6: output projection
  gemm_proj_kernel<<<dim3(kDIM / 128, NTOK / 128), 256, 0, stream>>>(obn, wprjb, out);
}
